// MGHDv2_61933428416919
// MI455X (gfx1250) — matrix operand order repaired and checked by reading, not compiled, not run on hardware
//
#include <hip/hip_runtime.h>
#include <hip/hip_bf16.h>

typedef __attribute__((ext_vector_type(16))) __bf16 v16bf;
typedef __attribute__((ext_vector_type(8)))  float  v8f;
typedef __attribute__((ext_vector_type(4)))  unsigned int u32x4;
typedef __attribute__((ext_vector_type(8)))  int i32x8;
typedef __attribute__((ext_vector_type(4)))  int i32x4;

#if defined(__has_builtin)
#if __has_builtin(__builtin_amdgcn_tensor_load_to_lds) && __has_builtin(__builtin_amdgcn_s_wait_tensorcnt)
#define HAVE_TDM 1
#endif
#endif
#ifndef HAVE_TDM
#define HAVE_TDM 0
#endif

#define NNODE 50000
#define NEDGE 400000
#define SSEQ  25000
#define DM    192
#define DI    384
#define DST8  80
#define DTR   12
#define XDBL  172          // DTR + 2*DST8
#define NITER 8

__device__ __forceinline__ __bf16 tobf(float f) {
  unsigned u = __builtin_bit_cast(unsigned, f);
  unsigned r = u + 0x7FFFu + ((u >> 16) & 1u);           // RNE
  unsigned short h = (unsigned short)(r >> 16);
  return __builtin_bit_cast(__bf16, h);
}
__device__ __forceinline__ float sigm(float x) { return 1.f / (1.f + __expf(-x)); }

// ---------------------------------------------------------------- small kernels
__global__ void k_cvt(const float* __restrict__ in, __bf16* __restrict__ out, int n) {
  int i = blockIdx.x * 256 + threadIdx.x;
  if (i < n) out[i] = tobf(in[i]);
}

// W is K x N row-major f32 -> Bt is N x K row-major bf16 (transposed weight)
__global__ void k_cvtT(const float* __restrict__ W, __bf16* __restrict__ Bt, int K, int N) {
  int i = blockIdx.x * 256 + threadIdx.x;
  if (i >= K * N) return;
  int n = i / K, k = i - n * K;
  Bt[i] = tobf(W[(size_t)k * N + n]);
}

__global__ void k_zero(float* __restrict__ p, int n) {
  int i = blockIdx.x * 256 + threadIdx.x;
  if (i < n) p[i] = 0.f;
}

__global__ void k_gcvt(const float* __restrict__ x, const int* __restrict__ idx,
                       __bf16* __restrict__ out, int rows, int cols) {
  int i = blockIdx.x * 256 + threadIdx.x;
  if (i >= rows * cols) return;
  int t = i / cols, c = i - t * cols;
  out[i] = tobf(x[(size_t)idx[t] * cols + c]);
}

__global__ void k_embed(const float* __restrict__ xn, const float* __restrict__ Wn,
                        const float* __restrict__ bn, const float* __restrict__ g,
                        const float* __restrict__ Wg, const float* __restrict__ bg,
                        float* __restrict__ x) {
  int i = blockIdx.x * 256 + threadIdx.x;
  if (i >= NNODE * DM) return;
  int n = i / DM, c = i - n * DM;
  float a = bn[c] + bg[c];
#pragma unroll
  for (int k = 0; k < 8; k++) a += xn[n * 8 + k] * Wn[k * DM + c] + g[k] * Wg[k * DM + c];
  x[i] = a;
}

__global__ void k_conv(const float* __restrict__ xz, const float* __restrict__ cw,
                       const float* __restrict__ cb, float* __restrict__ xcf,
                       __bf16* __restrict__ xcb) {
  int i = blockIdx.x * 256 + threadIdx.x;
  if (i >= SSEQ * DI) return;
  int t = i / DI, d = i - t * DI;
  float a = cb[d];
#pragma unroll
  for (int j = 0; j < 4; j++) {
    int tt = t - 3 + j;
    if (tt >= 0) a += xz[(size_t)tt * (2 * DI) + d] * cw[d * 4 + j];
  }
  float v = a * sigm(a);                                  // silu
  xcf[i] = v; xcb[i] = tobf(v);
}

__global__ void k_dt(const float* __restrict__ xdbl, const float* __restrict__ Wdt,
                     const float* __restrict__ bdt, float* __restrict__ dt) {
  int i = blockIdx.x * 256 + threadIdx.x;
  if (i >= SSEQ * DI) return;
  int t = i / DI, d = i - t * DI;
  float a = bdt[d];
#pragma unroll
  for (int r = 0; r < DTR; r++) a += xdbl[(size_t)t * XDBL + r] * Wdt[r * DI + d];
  dt[i] = (a > 20.f) ? a : __logf(1.f + __expf(a));       // softplus
}

__global__ void k_negA(const float* __restrict__ A_log, float* __restrict__ negA) {
  int i = blockIdx.x * 256 + threadIdx.x;
  if (i < DI * DST8) negA[i] = -__expf(A_log[i]);
}

// sequential SSM scan: one block, thread d owns channel d, 80-wide state in regs
__global__ void __launch_bounds__(DI) k_scan(const float* __restrict__ dt,
                                             const float* __restrict__ xdbl,
                                             const float* __restrict__ xc,
                                             const float* __restrict__ negA,
                                             const float* __restrict__ Dskip,
                                             float* __restrict__ ys) {
  int d = threadIdx.x;
  float s[DST8], An[DST8];
#pragma unroll
  for (int n = 0; n < DST8; n++) { s[n] = 0.f; An[n] = negA[d * DST8 + n]; }
  __shared__ float Bsh[DST8], Csh[DST8];
  float dsk = Dskip[d];
  for (int t = 0; t < SSEQ; t++) {
    if (d < DST8)            Bsh[d]        = xdbl[(size_t)t * XDBL + DTR + d];
    else if (d < 2 * DST8)   Csh[d - DST8] = xdbl[(size_t)t * XDBL + DTR + DST8 + (d - DST8)];
    __syncthreads();
    float dtv = dt[(size_t)t * DI + d];
    float xv  = xc[(size_t)t * DI + d];
    float dx  = dtv * xv;
    float acc = 0.f;
#pragma unroll
    for (int n = 0; n < DST8; n++) {
      s[n] = s[n] * __expf(dtv * An[n]) + dx * Bsh[n];
      acc += s[n] * Csh[n];
    }
    ys[(size_t)t * DI + d] = acc + dsk * xv;
    __syncthreads();
  }
}

__global__ void k_gate(const float* __restrict__ ys, const float* __restrict__ xz,
                       __bf16* __restrict__ yin) {
  int i = blockIdx.x * 256 + threadIdx.x;
  if (i >= SSEQ * DI) return;
  int t = i / DI, d = i - t * DI;
  float z = xz[(size_t)t * (2 * DI) + DI + d];
  yin[i] = tobf(ys[i] * (z * sigm(z)));
}

__global__ void __launch_bounds__(256) k_ln(const float* __restrict__ y,
                                            const float* __restrict__ g,
                                            const float* __restrict__ b,
                                            const int* __restrict__ seq_idx,
                                            float* __restrict__ x) {
  int t = blockIdx.x, c = threadIdx.x;
  __shared__ float red[256];
  float v = (c < DM) ? y[(size_t)t * DM + c] : 0.f;
  red[c] = v; __syncthreads();
  for (int s = 128; s > 0; s >>= 1) { if (c < s) red[c] += red[c + s]; __syncthreads(); }
  float mu = red[0] / DM; __syncthreads();
  red[c] = (c < DM) ? (v - mu) * (v - mu) : 0.f; __syncthreads();
  for (int s = 128; s > 0; s >>= 1) { if (c < s) red[c] += red[c + s]; __syncthreads(); }
  float var = red[0] / DM;
  if (c < DM) {
    float o = (v - mu) * rsqrtf(var + 1e-5f) * g[c] + b[c];
    x[(size_t)seq_idx[t] * DM + c] += o;
  }
}

__global__ void __launch_bounds__(256) k_colmean(const float* __restrict__ x,
                                                 float* __restrict__ mean) {
  int c = blockIdx.x, t = threadIdx.x;
  float s = 0.f;
  for (int i = t; i < NNODE; i += 256) s += x[(size_t)i * DM + c];
  __shared__ float red[256];
  red[t] = s; __syncthreads();
  for (int k = 128; k > 0; k >>= 1) { if (t < k) red[t] += red[t + k]; __syncthreads(); }
  if (t == 0) mean[c] = red[0] / (float)NNODE;
}

__global__ void __launch_bounds__(256) k_se(const float* __restrict__ mean,
                                            const float* __restrict__ W1, const float* __restrict__ b1,
                                            const float* __restrict__ W2, const float* __restrict__ b2,
                                            float* __restrict__ se) {
  __shared__ float hid[DTR];
  int t = threadIdx.x;
  if (t < DTR) {
    float a = b1[t];
    for (int c = 0; c < DM; c++) a += mean[c] * W1[c * DTR + t];
    hid[t] = fmaxf(a, 0.f);
  }
  __syncthreads();
  if (t < DM) {
    float a = b2[t];
#pragma unroll
    for (int r = 0; r < DTR; r++) a += hid[r] * W2[r * DM + t];
    se[t] = sigm(a);
  }
}

__global__ void k_scale(const float* __restrict__ x, const float* __restrict__ se,
                        float* __restrict__ h) {
  int i = blockIdx.x * 256 + threadIdx.x;
  if (i < NNODE * DM) h[i] = x[i] * se[i % DM];
}

__global__ void k_logits(const float* __restrict__ h, const float* __restrict__ Wo,
                         const float* __restrict__ bo, const unsigned char* __restrict__ mask,
                         float* __restrict__ out, int out_size) {
  int i = blockIdx.x * 256 + threadIdx.x;
  if (i < NNODE * 2) {
    int n = i / 2, j = i - n * 2;
    float a = bo[j];
    for (int c = 0; c < DM; c++) a += h[(size_t)n * DM + c] * Wo[c * 2 + j];
    out[i] = a;
  }
  if (i < NNODE && out_size >= NNODE * 2 + NNODE) out[NNODE * 2 + i] = mask[i] ? 1.f : 0.f;
}

// ---------------------------------------------------------------- TDM helper
// 2D tile load: rows x 32 bf16 elements, row stride `stride` elements, into LDS
// at byte offset lds_off with 48-element (96B) padded rows:
// pad_interval code 3 = 16 DWORDs (64B = 32 bf16), pad_amount code 7 = 8 DWORDs (32B).
#if HAVE_TDM
__device__ __forceinline__ void tdm_load_2d(unsigned lds_off, const void* gaddr,
                                            unsigned rows, unsigned long long tdim1,
                                            unsigned long long stride) {
  unsigned long long ga = (unsigned long long)(uintptr_t)gaddr;
  u32x4 g0;
  g0[0] = 1u;                                            // count=1, user mode
  g0[1] = lds_off;                                       // lds_addr
  g0[2] = (unsigned)(ga & 0xFFFFFFFFu);                  // global_addr[31:0]
  g0[3] = (unsigned)((ga >> 32) & 0x1FFFFFFu) | (2u << 30);  // addr[56:32] | type=2
  unsigned long long td0 = 32ull;
  i32x8 g1;
  g1[0] = (int)((1u << 16) | (1u << 20) | (3u << 22) | (7u << 25)); // 2B elems, pad on
  g1[1] = (int)((td0 & 0xFFFFull) << 16);
  g1[2] = (int)(((td0 >> 16) & 0xFFFFull) | ((tdim1 & 0xFFFFull) << 16));
  g1[3] = (int)(((tdim1 >> 16) & 0xFFFFull) | (32u << 16));          // tile_dim0=32
  g1[4] = (int)(rows & 0xFFFFu);                                     // tile_dim1
  g1[5] = (int)(stride & 0xFFFFFFFFull);
  g1[6] = (int)((stride >> 32) & 0xFFFFull);
  g1[7] = 0;
  i32x4 z4 = {0, 0, 0, 0};
#if defined(__clang_major__) && (__clang_major__ >= 23)
  i32x8 z8 = {0, 0, 0, 0, 0, 0, 0, 0};
  __builtin_amdgcn_tensor_load_to_lds(g0, g1, z4, z4, z8, 0);
#else
  __builtin_amdgcn_tensor_load_to_lds(g0, g1, z4, z4, 0);
#endif
}
#endif

// ---------------------------------------------------------------- WMMA GEMM
// C[M,N] = act(A[M,K] * Bt^T [K,N] + bias), Bt is N x K bf16 (pre-transposed weight)
// A elem (m,k): k<ksplit -> p0[(idx0?idx0[m]:m)*lda0 + k]  else p1[(idx1?idx1[m]:m)*lda1 + k-ksplit]
// OUT: 0=f32 store, 1=bf16 store, 2=atomicAdd into Cf[sidx[m]*ldc+n] (masked)
// ACT: 0=none, 1=relu, 2=tanh.  TDMV: use Tensor Data Mover for A (requires idx==null path ok)
template <int ACT, int OUT, int TDMV>
__global__ void __launch_bounds__(256) k_gemm(
    const __bf16* __restrict__ p0, int lda0,
    const __bf16* __restrict__ p1, int lda1, int ksplit,
    const int* __restrict__ idx0, const int* __restrict__ idx1,
    const __bf16* __restrict__ Bt, int ldb,
    const float* __restrict__ bias,
    float* __restrict__ Cf, __bf16* __restrict__ Cb, int ldc,
    const int* __restrict__ sidx, const unsigned char* __restrict__ smask,
    int M, int N, int K) {
  constexpr int UT = TDMV && HAVE_TDM;
  __shared__ __align__(16) __bf16 As[UT ? 2 : 1][128][48];  // [m][k]
  __shared__ __align__(16) __bf16 Bs[128][48];              // [n][k]
  const int tid  = threadIdx.x;
  const int lane = tid & 31, wid = tid >> 5;
  const int wm = wid >> 1, wn = wid & 1;        // 4 x 2 waves -> 128 x 128 tile
  const int lrow = lane & 15;
  const bool hi = lane >= 16;
  const int m0 = blockIdx.y * 128;
  const int n0 = blockIdx.x * 128;

  const v8f vz = {0.f, 0.f, 0.f, 0.f, 0.f, 0.f, 0.f, 0.f};
  v8f acc[2][4];
#pragma unroll
  for (int a = 0; a < 2; a++)
#pragma unroll
    for (int b = 0; b < 4; b++) acc[a][b] = vz;

  const int arow  = tid >> 1;             // 0..127 (A row / B col)
  const int apart = (tid & 1) << 4;       // 0 or 16 (k sub-chunk)
  const uint4 uz = {0u, 0u, 0u, 0u};

  // per-thread loop-invariant A row bases (manual path only)
  const __bf16 *base0 = p0, *base1 = p1;
  bool mval = true;
  if (!UT) {
    int m = m0 + arow;
    mval = m < M;
    int mc = mval ? m : (M - 1);
    int r0 = idx0 ? idx0[mc] : mc;
    int r1 = idx1 ? idx1[mc] : mc;
    base0 = p0 + (size_t)r0 * lda0;
    base1 = p1 + (size_t)r1 * lda1 - ksplit;
  }
  const bool nval = (n0 + arow) < N;
  const __bf16* bbase = Bt + (size_t)(n0 + arow) * ldb;

  uint4 ra0 = uz, ra1 = uz, rb0 = uz, rb1 = uz;
  auto loadA = [&](int k0) {
    int gk = k0 + apart;
    const __bf16* s = ((gk < ksplit) ? base0 : base1) + gk;
    ra0 = mval ? *(const uint4*)s : uz;
    ra1 = mval ? *(const uint4*)(s + 8) : uz;
  };
  auto loadB = [&](int k0) {
    const __bf16* s = bbase + k0 + apart;
    rb0 = nval ? *(const uint4*)s : uz;
    rb1 = nval ? *(const uint4*)(s + 8) : uz;
  };
#if HAVE_TDM
  auto tdmIssue = [&](int k0, int b) {
    const __bf16* gsrc; int stride;
    if (k0 < ksplit) { gsrc = p0 + (size_t)m0 * lda0 + k0;            stride = lda0; }
    else             { gsrc = p1 + (size_t)m0 * lda1 + (k0 - ksplit); stride = lda1; }
    unsigned lds = (unsigned)(uintptr_t)&As[b][0][0];
    tdm_load_2d(lds, gsrc, 128u, (unsigned long long)(M - m0), (unsigned long long)stride);
  };
#endif

  // prologue
#if HAVE_TDM
  if (UT) { if (wid == 0) tdmIssue(0, 0); }
  else loadA(0);
#else
  loadA(0);
#endif
  loadB(0);

  int buf = 0;
  for (int k0 = 0; k0 < K; k0 += 32) {
    const bool nxt = (k0 + 32) < K;
    if (!UT) {
      *(uint4*)&As[0][arow][apart]     = ra0;
      *(uint4*)&As[0][arow][apart + 8] = ra1;
    }
#if HAVE_TDM
    else if (wid == 0) {
      if (nxt) { tdmIssue(k0 + 32, buf ^ 1); __builtin_amdgcn_s_wait_tensorcnt(1); }
      else     { __builtin_amdgcn_s_wait_tensorcnt(0); }
    }
#endif
    *(uint4*)&Bs[arow][apart]     = rb0;
    *(uint4*)&Bs[arow][apart + 8] = rb1;
    __syncthreads();
    if (nxt) { if (!UT) loadA(k0 + 32); loadB(k0 + 32); }

    union F { uint4 u[2]; v16bf v; };
    F a[2], b[4];
    const __bf16 (*Ac)[48] = As[UT ? buf : 0];
    const int ka = hi ? 8 : 0;   // A frag: lanes0-15 K=0..7,16..23 ; lanes16-31 K=8..15,24..31
#pragma unroll
    for (int mi = 0; mi < 2; mi++) {
      const __bf16* ap = &Ac[wm * 32 + mi * 16 + lrow][ka];
      a[mi].u[0] = *(const uint4*)ap;
      a[mi].u[1] = *(const uint4*)(ap + 16);
    }
    const int kb = hi ? 8 : 0;   // B frag: lanes0-15 K=0..7,16..23 ; lanes16-31 K=8..15,24..31
#pragma unroll
    for (int ni = 0; ni < 4; ni++) {
      const __bf16* bp = &Bs[wn * 64 + ni * 16 + lrow][kb];
      b[ni].u[0] = *(const uint4*)bp;
      b[ni].u[1] = *(const uint4*)(bp + 16);
    }
#pragma unroll
    for (int mi = 0; mi < 2; mi++)
#pragma unroll
      for (int ni = 0; ni < 4; ni++)
        acc[mi][ni] = __builtin_amdgcn_wmma_f32_16x16x32_bf16(
            false, a[mi].v, false, b[ni].v, (short)0, acc[mi][ni], false, false);
    __syncthreads();
    buf ^= 1;
  }

  // epilogue: D VGPR v -> row (v or v+8), col = lane%16
#pragma unroll
  for (int mi = 0; mi < 2; mi++)
#pragma unroll
    for (int ni = 0; ni < 4; ni++)
#pragma unroll
      for (int v = 0; v < 8; v++) {
        int gm = m0 + wm * 32 + mi * 16 + (hi ? v + 8 : v);
        int gn = n0 + wn * 64 + ni * 16 + lrow;
        if (gm < M && gn < N) {
          float val = acc[mi][ni][v];
          if (bias) val += bias[gn];
          if (ACT == 1) val = fmaxf(val, 0.f);
          if (ACT == 2) val = tanhf(val);
          if (OUT == 0)      Cf[(size_t)gm * ldc + gn] = val;
          else if (OUT == 1) Cb[(size_t)gm * ldc + gn] = tobf(val);
          else if (!smask || smask[gm]) atomicAdd(&Cf[(size_t)sidx[gm] * ldc + gn], val);
        }
      }
}

// ---------------------------------------------------------------- launcher
extern "C" void kernel_launch(void* const* d_in, const int* in_sizes, int n_in,
                              void* d_out, int out_size, void* d_ws, size_t ws_size,
                              hipStream_t stream) {
  (void)in_sizes; (void)n_in; (void)ws_size;
  const float* x_nodes = (const float*)d_in[0];
  const float* g_token = (const float*)d_in[2];
  const float* Wn = (const float*)d_in[3];  const float* bn = (const float*)d_in[4];
  const float* Wg = (const float*)d_in[7];  const float* bg = (const float*)d_in[8];
  const float* Win = (const float*)d_in[9];
  const float* conv_w = (const float*)d_in[10]; const float* conv_b = (const float*)d_in[11];
  const float* Wx = (const float*)d_in[12]; const float* Wdt = (const float*)d_in[13];
  const float* bdt = (const float*)d_in[14]; const float* A_log = (const float*)d_in[15];
  const float* Dskip = (const float*)d_in[16]; const float* Wout = (const float*)d_in[17];
  const float* ln_g = (const float*)d_in[18]; const float* ln_b = (const float*)d_in[19];
  const float* Wse1 = (const float*)d_in[20]; const float* bse1 = (const float*)d_in[21];
  const float* Wse2 = (const float*)d_in[22]; const float* bse2 = (const float*)d_in[23];
  const float* Wm1 = (const float*)d_in[24]; const float* bm1 = (const float*)d_in[25];
  const float* Wm2 = (const float*)d_in[26]; const float* bm2 = (const float*)d_in[27];
  const float* Wu = (const float*)d_in[28];  const float* bu = (const float*)d_in[29];
  const float* Wo = (const float*)d_in[30];  const float* bo = (const float*)d_in[31];
  const int* edge_index = (const int*)d_in[32];
  const unsigned char* node_mask = (const unsigned char*)d_in[33];
  const unsigned char* edge_mask = (const unsigned char*)d_in[34];
  const int* seq_idx = (const int*)d_in[35];
  const int* src = edge_index;
  const int* dst = edge_index + NEDGE;

  char* base = (char*)d_ws; size_t off = 0;
  auto alloc = [&](size_t bytes) -> void* { void* p = base + off; off = (off + bytes + 255) & ~(size_t)255; return p; };

  __bf16* WinT  = (__bf16*)alloc((size_t)DM * 2 * DI * 2);   // [768][192]
  __bf16* WxT   = (__bf16*)alloc((size_t)DI * XDBL * 2);     // [172][384]
  __bf16* WoutT = (__bf16*)alloc((size_t)DI * DM * 2);       // [192][384]
  __bf16* Wm1T  = (__bf16*)alloc((size_t)2 * DM * DM * 2);   // [192][384]
  __bf16* Wm2T  = (__bf16*)alloc((size_t)DM * DM * 2);       // [192][192]
  __bf16* WuT   = (__bf16*)alloc((size_t)2 * DM * DM * 2);   // [192][384]
  float*  negA  = (float*)alloc((size_t)DI * DST8 * 4);
  float*  xbuf  = (float*)alloc((size_t)NNODE * DM * 4);
  __bf16* u_bf  = (__bf16*)alloc((size_t)SSEQ * DM * 2);
  float*  xz    = (float*)alloc((size_t)SSEQ * 2 * DI * 4);
  float*  xcf   = (float*)alloc((size_t)SSEQ * DI * 4);
  __bf16* xcb   = (__bf16*)alloc((size_t)SSEQ * DI * 2);
  float*  xdbl  = (float*)alloc((size_t)SSEQ * XDBL * 4);
  float*  dtb   = (float*)alloc((size_t)SSEQ * DI * 4);
  float*  ys    = (float*)alloc((size_t)SSEQ * DI * 4);
  __bf16* yin   = (__bf16*)alloc((size_t)SSEQ * DI * 2);
  float*  y192  = (float*)alloc((size_t)SSEQ * DM * 4);
  float*  meanb = (float*)alloc(DM * 4);
  float*  seb   = (float*)alloc(DM * 4);
  float*  hA    = (float*)alloc((size_t)NNODE * DM * 4);
  float*  hB    = (float*)alloc((size_t)NNODE * DM * 4);
  __bf16* hbf   = (__bf16*)alloc((size_t)NNODE * DM * 2);
  float*  agg   = (float*)alloc((size_t)NNODE * DM * 4);
  __bf16* aggbf = (__bf16*)alloc((size_t)NNODE * DM * 2);
  __bf16* m1bf  = (__bf16*)alloc((size_t)NEDGE * DM * 2);

  auto cdiv = [](long long a, long long b) { return (int)((a + b - 1) / b); };
  dim3 blk(256);

  // weights -> transposed bf16
  k_cvtT<<<cdiv(DM * 2 * DI, 256), blk, 0, stream>>>(Win, WinT, DM, 2 * DI);
  k_cvtT<<<cdiv(DI * XDBL, 256), blk, 0, stream>>>(Wx, WxT, DI, XDBL);
  k_cvtT<<<cdiv(DI * DM, 256), blk, 0, stream>>>(Wout, WoutT, DI, DM);
  k_cvtT<<<cdiv(2 * DM * DM, 256), blk, 0, stream>>>(Wm1, Wm1T, 2 * DM, DM);
  k_cvtT<<<cdiv(DM * DM, 256), blk, 0, stream>>>(Wm2, Wm2T, DM, DM);
  k_cvtT<<<cdiv(2 * DM * DM, 256), blk, 0, stream>>>(Wu, WuT, 2 * DM, DM);
  k_negA<<<cdiv(DI * DST8, 256), blk, 0, stream>>>(A_log, negA);

  // embedding
  k_embed<<<cdiv((long long)NNODE * DM, 256), blk, 0, stream>>>(x_nodes, Wn, bn, g_token, Wg, bg, xbuf);

  // ---- Mamba
  k_gcvt<<<cdiv((long long)SSEQ * DM, 256), blk, 0, stream>>>(xbuf, seq_idx, u_bf, SSEQ, DM);
  k_gemm<0, 0, 1><<<dim3(cdiv(2 * DI, 128), cdiv(SSEQ, 128)), blk, 0, stream>>>(
      u_bf, DM, u_bf, DM, DM, nullptr, nullptr, WinT, DM, nullptr,
      xz, nullptr, 2 * DI, nullptr, nullptr, SSEQ, 2 * DI, DM);
  k_conv<<<cdiv((long long)SSEQ * DI, 256), blk, 0, stream>>>(xz, conv_w, conv_b, xcf, xcb);
  k_gemm<0, 0, 1><<<dim3(cdiv(XDBL, 128), cdiv(SSEQ, 128)), blk, 0, stream>>>(
      xcb, DI, xcb, DI, DI, nullptr, nullptr, WxT, DI, nullptr,
      xdbl, nullptr, XDBL, nullptr, nullptr, SSEQ, XDBL, DI);
  k_dt<<<cdiv((long long)SSEQ * DI, 256), blk, 0, stream>>>(xdbl, Wdt, bdt, dtb);
  k_scan<<<1, DI, 0, stream>>>(dtb, xdbl, xcf, negA, Dskip, ys);
  k_gate<<<cdiv((long long)SSEQ * DI, 256), blk, 0, stream>>>(ys, xz, yin);
  k_gemm<0, 0, 1><<<dim3(cdiv(DM, 128), cdiv(SSEQ, 128)), blk, 0, stream>>>(
      yin, DI, yin, DI, DI, nullptr, nullptr, WoutT, DI, nullptr,
      y192, nullptr, DM, nullptr, nullptr, SSEQ, DM, DI);
  k_ln<<<SSEQ, blk, 0, stream>>>(y192, ln_g, ln_b, seq_idx, xbuf);

  // ---- SE
  k_colmean<<<DM, blk, 0, stream>>>(xbuf, meanb);
  k_se<<<1, blk, 0, stream>>>(meanb, Wse1, bse1, Wse2, bse2, seb);
  k_scale<<<cdiv((long long)NNODE * DM, 256), blk, 0, stream>>>(xbuf, seb, hA);

  // ---- message passing
  float* hcur = hA; float* hnext = hB;
  for (int it = 0; it < NITER; ++it) {
    k_cvt<<<cdiv((long long)NNODE * DM, 256), blk, 0, stream>>>(hcur, hbf, NNODE * DM);
    k_zero<<<cdiv((long long)NNODE * DM, 256), blk, 0, stream>>>(agg, NNODE * DM);
    // layer1: gather [h[src], h[dst]] -> relu -> bf16   (manual A path: gathered rows)
    k_gemm<1, 1, 0><<<dim3(cdiv(DM, 128), cdiv(NEDGE, 128)), blk, 0, stream>>>(
        hbf, DM, hbf, DM, DM, src, dst, Wm1T, 2 * DM, bm1,
        nullptr, m1bf, DM, nullptr, nullptr, NEDGE, DM, 2 * DM);
    // layer2: + bias, masked atomic scatter-add into agg[dst]   (TDM A path)
    k_gemm<0, 2, 1><<<dim3(cdiv(DM, 128), cdiv(NEDGE, 128)), blk, 0, stream>>>(
        m1bf, DM, m1bf, DM, DM, nullptr, nullptr, Wm2T, DM, bm2,
        agg, nullptr, DM, dst, edge_mask, NEDGE, DM, DM);
    k_cvt<<<cdiv((long long)NNODE * DM, 256), blk, 0, stream>>>(agg, aggbf, NNODE * DM);
    // node update: tanh([h, agg] @ Wu + bu)   (TDM A path, two-segment concat)
    k_gemm<2, 0, 1><<<dim3(cdiv(DM, 128), cdiv(NNODE, 128)), blk, 0, stream>>>(
        hbf, DM, aggbf, DM, DM, nullptr, nullptr, WuT, 2 * DM, bu,
        hnext, nullptr, DM, nullptr, nullptr, NNODE, DM, 2 * DM);
    float* tmp = hcur; hcur = hnext; hnext = tmp;
  }

  k_logits<<<cdiv((long long)NNODE * 2, 256), blk, 0, stream>>>(
      hcur, Wo, bo, node_mask, (float*)d_out, out_size);
}
